// SpatialCrossAttentionLite_62517543960664
// MI455X (gfx1250) — matrix operand order repaired and checked by reading, not compiled, not run on hardware
//
#include <hip/hip_runtime.h>
#include <hip/hip_bf16.h>
#include <stdint.h>

// ---- problem constants ----
#define CAMS   6
#define NQ     6400
#define EMBED  256
#define HEADS  8
#define LEVELS 4
#define POINTS 8
#define DEPTH  4
#define DH     32
#define FLEN   14960
#define VROWS  (FLEN * CAMS)                    // 89760 value rows
#define SO_COLS (HEADS * LEVELS * POINTS * 2)   // 512
#define AW_COLS (HEADS * LEVELS * POINTS)       // 256

typedef __attribute__((ext_vector_type(16))) __bf16        v16bf;
typedef __attribute__((ext_vector_type(8)))  float         v8f;
typedef __attribute__((ext_vector_type(4)))  unsigned int  u32x4;
typedef __attribute__((ext_vector_type(4)))  int           i32x4;

#if __has_builtin(__builtin_amdgcn_global_load_async_to_lds_b128)
#define HAVE_ASYNC_LDS 1
typedef __attribute__((address_space(1))) i32x4* as1_i32x4p;   // global
typedef __attribute__((address_space(3))) i32x4* as3_i32x4p;   // LDS
#endif

__device__ __forceinline__ unsigned short f32_to_bf16(float f) {
    unsigned int u = __float_as_uint(f);
    u += 0x7FFFu + ((u >> 16) & 1u);          // round-to-nearest-even
    return (unsigned short)(u >> 16);
}

__device__ __forceinline__ void wait_asynccnt0() {
#if __has_builtin(__builtin_amdgcn_s_wait_asynccnt)
    __builtin_amdgcn_s_wait_asynccnt(0);
#else
    asm volatile("s_wait_asynccnt 0x0" ::: "memory");
#endif
}

// ---------------------------------------------------------------------------
// Pack a row-major f32 KxN weight matrix into the bf16 WMMA B-fragment layout:
// Bp[((kt*Ntiles + nt)*32 + lane)*16 + i] = W[kt*32 + (lane<16?0:8)+(i&7)+16*(i>>3)][nt*16 + (lane&15)]
// so each wave B-fragment is one 32-byte contiguous per-lane load.
// ---------------------------------------------------------------------------
__global__ void pack_b_kernel(const float* __restrict__ W,
                              unsigned short* __restrict__ Bp, int K, int N) {
    int t = blockIdx.x * blockDim.x + threadIdx.x;
    if (t >= K * N) return;
    int i    = t & 15;
    int lane = (t >> 4) & 31;
    int rest = t >> 9;
    int Ntiles = N >> 4;
    int nt = rest % Ntiles;
    int kt = rest / Ntiles;
    int Kidx = kt * 32 + ((lane < 16) ? 0 : 8) + (i & 7) + ((i >> 3) << 4);
    int Nidx = nt * 16 + (lane & 15);
    Bp[t] = f32_to_bf16(W[(size_t)Kidx * N + Nidx]);
}

__global__ void add_cast_bf16_kernel(const float* __restrict__ a,
                                     const float* __restrict__ b,
                                     unsigned short* __restrict__ o, int n) {
    int t = blockIdx.x * blockDim.x + threadIdx.x;
    if (t < n) o[t] = f32_to_bf16(a[t] + b[t]);
}

__global__ void cast_bf16_kernel(const float* __restrict__ a,
                                 unsigned short* __restrict__ o, int n) {
    int t = blockIdx.x * blockDim.x + threadIdx.x;
    if (t < n) o[t] = f32_to_bf16(a[t]);
}

// ---------------------------------------------------------------------------
// bf16 WMMA GEMM:  C[M,N] = A[M,K](bf16) x B(packed bf16) + bias (+ resid)
// Block = 256 threads = 8 waves arranged 2(M) x 4(N); each wave computes FOUR
// 16x16 C tiles sharing one A fragment -> 4 WMMAs per A-frag load (A-reuse).
// Block tile = 32 rows x 256 cols. grid = (N/256, M/32).
// M%32==0, N%256==0, K%32==0 for all calls -> no divergence, EXEC all ones.
// ---------------------------------------------------------------------------
__global__ void __launch_bounds__(256)
wmma_gemm_kernel(const unsigned short* __restrict__ A,
                 const unsigned short* __restrict__ Bp,
                 const float* __restrict__ bias,
                 const float* __restrict__ resid,
                 float* __restrict__ C, int M, int N, int K) {
    const int lane = threadIdx.x & 31;
    const int wave = threadIdx.x >> 5;
    const int wm = wave >> 2;                   // 0..1  (M)
    const int wn = wave & 3;                    // 0..3  (N)
    const int mtile  = blockIdx.y * 2 + wm;
    const int ntile0 = blockIdx.x * 16 + wn * 4;
    const int Ntiles = N >> 4;

    const int arow  = mtile * 16 + (lane & 15);
    const int kbase = (lane < 16) ? 0 : 8;      // ISA 16-bit A-matrix half-K select
    const unsigned short* aptr = A + (size_t)arow * K;

    union Frag { v16bf v; u32x4 q[2]; };
    v8f acc[4] = {{}, {}, {}, {}};

    const int Kt = K >> 5;
    for (int kt = 0; kt < Kt; ++kt) {
        const int k0 = kt << 5;
        Frag a;
        // A 16x32 bf16 fragment: lane holds K = kbase..kbase+7 and kbase+16..kbase+23
        a.q[0] = *(const u32x4*)(aptr + k0 + kbase);
        a.q[1] = *(const u32x4*)(aptr + k0 + kbase + 16);
        __builtin_prefetch(aptr + k0 + 32, 0, 1);          // global_prefetch_b8
        const unsigned short* bbase =
            Bp + (((size_t)kt * Ntiles + ntile0) * 32 + lane) * 16;
#pragma unroll
        for (int j = 0; j < 4; ++j) {
            Frag b;   // packed: consecutive ntiles are +512 shorts apart
            const unsigned short* bp = bbase + (size_t)j * 512;
            b.q[0] = *(const u32x4*)(bp);
            b.q[1] = *(const u32x4*)(bp + 8);
            acc[j] = __builtin_amdgcn_wmma_f32_16x16x32_bf16(
                         false, a.v, false, b.v, (short)0, acc[j], false, false);
        }
    }

    // C/D layout: VGPR r -> row (lane<16 ? r : 8+r), col = lane&15
    const int r0 = mtile * 16 + ((lane < 16) ? 0 : 8);
#pragma unroll
    for (int j = 0; j < 4; ++j) {
        const int col = (ntile0 + j) * 16 + (lane & 15);
        const float bb = bias ? bias[col] : 0.0f;
#pragma unroll
        for (int r = 0; r < 8; ++r) {
            size_t o = (size_t)(r0 + r) * N + col;
            float v = acc[j][r] + bb;
            if (resid) v += resid[o];
            C[o] = v;
        }
    }
}

// ---------------------------------------------------------------------------
// Sampling + cam aggregation. Block = 256 (8 waves), one block per query,
// one wave per head, lanes span the 32-wide head slice (DH).
// Stages the query's 512-float offset row into LDS via the CDNA5 async
// global->LDS path (ASYNCcnt), fuses softmax via shfl + LDS broadcast,
// bilinear MSDA over 4 levels x 8 points, cam masking, 1/count normalization.
// ---------------------------------------------------------------------------
__global__ void __launch_bounds__(256)
sample_kernel(const float* __restrict__ vproj,       // (VROWS, 256) row = f*CAMS + c
              const float* __restrict__ off_logits,  // (NQ, 512)
              const float* __restrict__ aw_logits,   // (NQ, 256)
              const float* __restrict__ refpts,      // (CAMS, 1, NQ, DEPTH, 2)
              const unsigned char* __restrict__ bev_mask, // (CAMS, 1, NQ, DEPTH) bool
              unsigned short* __restrict__ slots_bf16) {  // (NQ, 256)
    const int q    = blockIdx.x;
    const int h    = threadIdx.x >> 5;
    const int lane = threadIdx.x & 31;

    __shared__ __align__(16) float s_off[SO_COLS];      // this query's offsets
    __shared__ float s_aw[HEADS][LEVELS * POINTS];

    // ---- stage offset row (512 floats) into LDS ----
    const float* goff = off_logits + (size_t)q * SO_COLS;
    if (threadIdx.x < (SO_COLS / 4)) {                   // 128 lanes x 16B
#ifdef HAVE_ASYNC_LDS
        __builtin_amdgcn_global_load_async_to_lds_b128(
            (as1_i32x4p)(goff + threadIdx.x * 4),
            (as3_i32x4p)(&s_off[threadIdx.x * 4]),
            0, 0);
#else
        ((u32x4*)s_off)[threadIdx.x] = ((const u32x4*)goff)[threadIdx.x];
#endif
    }

    // ---- softmax over the 32 (level,point) logits; lane == (l*8+p) ----
    float x = aw_logits[(size_t)q * AW_COLS + h * 32 + lane];
    float m = x;
#pragma unroll
    for (int o = 16; o > 0; o >>= 1) m = fmaxf(m, __shfl_xor(m, o, 32));
    float e = __expf(x - m);
    float s = e;
#pragma unroll
    for (int o = 16; o > 0; o >>= 1) s += __shfl_xor(s, o, 32);
    s_aw[h][lane] = e / s;

#ifdef HAVE_ASYNC_LDS
    wait_asynccnt0();
#endif
    __syncthreads();

    const int   Wi[LEVELS]  = {176, 88, 44, 22};
    const int   Hi[LEVELS]  = {64, 32, 16, 8};
    const int   ls[LEVELS]  = {0, 11264, 14080, 14784};
    const float Wf[LEVELS]  = {176.f, 88.f, 44.f, 22.f};
    const float Hf[LEVELS]  = {64.f, 32.f, 16.f, 8.f};

    const float* offh = &s_off[h * (LEVELS * POINTS * 2)];

    float acc = 0.0f;
    int   cnt = 0;
    for (int c = 0; c < CAMS; ++c) {
        // 4 depth bools in one aligned u32; wave-uniform branch
        unsigned int mw = *(const unsigned int*)(bev_mask + ((size_t)c * NQ + q) * DEPTH);
        if (mw == 0u) continue;
        ++cnt;
        const float* refb = refpts + ((size_t)c * NQ + q) * (DEPTH * 2);
#pragma unroll
        for (int l = 0; l < LEVELS; ++l) {
            const int wl = Wi[l], hl = Hi[l], lsl = ls[l];
#pragma unroll
            for (int p = 0; p < POINTS; ++p) {
                const int d = p & 3;                       // p = pa*DEPTH + d
                const float ox = offh[l * 16 + p * 2 + 0];
                const float oy = offh[l * 16 + p * 2 + 1];
                // (ref + off/norm)*shape - 0.5 == ref*shape + off - 0.5
                const float xx = refb[d * 2 + 0] * Wf[l] + ox - 0.5f;
                const float yy = refb[d * 2 + 1] * Hf[l] + oy - 0.5f;
                const float x0 = floorf(xx), y0 = floorf(yy);
                const int xi = (int)x0, yi = (int)y0;
                const float fx = xx - x0, fy = yy - y0;
                const float aw = s_aw[h][l * 8 + p];
                const float w00 = (1.f - fx) * (1.f - fy) * aw;
                const float w10 = fx * (1.f - fy) * aw;
                const float w01 = (1.f - fx) * fy * aw;
                const float w11 = fx * fy * aw;
#pragma unroll
                for (int cr = 0; cr < 4; ++cr) {
                    const int X = xi + (cr & 1);
                    const int Y = yi + (cr >> 1);
                    const float w = (cr == 0) ? w00 : (cr == 1) ? w10 : (cr == 2) ? w01 : w11;
                    if (X >= 0 && X < wl && Y >= 0 && Y < hl) {   // wave-uniform
                        size_t row = ((size_t)(lsl + Y * wl + X) * CAMS + c) * EMBED;
                        acc += w * vproj[row + h * DH + lane];
                    }
                }
            }
        }
    }
    const float scale = 1.0f / fmaxf((float)cnt, 1.0f);
    slots_bf16[(size_t)q * EMBED + h * DH + lane] = f32_to_bf16(acc * scale);
}

// ---------------------------------------------------------------------------
extern "C" void kernel_launch(void* const* d_in, const int* in_sizes, int n_in,
                              void* d_out, int out_size, void* d_ws, size_t ws_size,
                              hipStream_t stream) {
    (void)in_sizes; (void)n_in; (void)out_size; (void)ws_size;

    const float* query     = (const float*)d_in[0];
    /* d_in[1] = key (unused by reference) */
    const float* value     = (const float*)d_in[2];
    const float* query_pos = (const float*)d_in[3];
    const float* refpts    = (const float*)d_in[4];
    const unsigned char* bev_mask = (const unsigned char*)d_in[5];
    const float* Wv   = (const float*)d_in[8];
    const float* bv   = (const float*)d_in[9];
    const float* Wso  = (const float*)d_in[10];
    const float* bso  = (const float*)d_in[11];
    const float* Waw  = (const float*)d_in[12];
    const float* baw  = (const float*)d_in[13];
    const float* Wout = (const float*)d_in[14];
    const float* bout = (const float*)d_in[15];
    float* out = (float*)d_out;

    // ---- workspace carve-up (256B aligned) ----
    char* ws = (char*)d_ws;
    size_t off = 0;
    auto carve = [&](size_t bytes) { char* p = ws + off; off = (off + bytes + 255) & ~(size_t)255; return p; };
    unsigned short* v_bf16    = (unsigned short*)carve((size_t)VROWS * EMBED * 2);   // 46 MB
    unsigned short* q_bf16    = (unsigned short*)carve((size_t)NQ * EMBED * 2);
    unsigned short* Wv_p      = (unsigned short*)carve((size_t)EMBED * EMBED * 2);
    unsigned short* Wso_p     = (unsigned short*)carve((size_t)EMBED * SO_COLS * 2);
    unsigned short* Waw_p     = (unsigned short*)carve((size_t)EMBED * AW_COLS * 2);
    unsigned short* Wout_p    = (unsigned short*)carve((size_t)EMBED * EMBED * 2);
    float*          vproj     = (float*)carve((size_t)VROWS * EMBED * 4);            // 92 MB
    float*          off_log   = (float*)carve((size_t)NQ * SO_COLS * 4);
    float*          aw_log    = (float*)carve((size_t)NQ * AW_COLS * 4);
    unsigned short* slots_b16 = (unsigned short*)carve((size_t)NQ * EMBED * 2);

    // ---- 1) pack weights into WMMA B-fragment layout ----
    {
        int n;
        n = EMBED * EMBED;   pack_b_kernel<<<(n + 255) / 256, 256, 0, stream>>>(Wv,   Wv_p,   EMBED, EMBED);
        n = EMBED * SO_COLS; pack_b_kernel<<<(n + 255) / 256, 256, 0, stream>>>(Wso,  Wso_p,  EMBED, SO_COLS);
        n = EMBED * AW_COLS; pack_b_kernel<<<(n + 255) / 256, 256, 0, stream>>>(Waw,  Waw_p,  EMBED, AW_COLS);
        n = EMBED * EMBED;   pack_b_kernel<<<(n + 255) / 256, 256, 0, stream>>>(Wout, Wout_p, EMBED, EMBED);
    }

    // ---- 2) q = query + query_pos -> bf16 ; value -> bf16 ----
    {
        int nq = NQ * EMBED;
        add_cast_bf16_kernel<<<(nq + 255) / 256, 256, 0, stream>>>(query, query_pos, q_bf16, nq);
        int nv = VROWS * EMBED;
        cast_bf16_kernel<<<(nv + 255) / 256, 256, 0, stream>>>(value, v_bf16, nv);
    }

    // ---- 3) WMMA GEMMs (block tile 32x256) ----
    // v_proj = value @ Wv + bv        : (89760 x 256) x (256 x 256)
    wmma_gemm_kernel<<<dim3(EMBED / 256, VROWS / 32), 256, 0, stream>>>(
        v_bf16, Wv_p, bv, nullptr, vproj, VROWS, EMBED, EMBED);
    // off_logits = q @ Wso + bso      : (6400 x 256) x (256 x 512)  (shared across cams)
    wmma_gemm_kernel<<<dim3(SO_COLS / 256, NQ / 32), 256, 0, stream>>>(
        q_bf16, Wso_p, bso, nullptr, off_log, NQ, SO_COLS, EMBED);
    // aw_logits = q @ Waw + baw       : (6400 x 256) x (256 x 256)  (shared across cams)
    wmma_gemm_kernel<<<dim3(AW_COLS / 256, NQ / 32), 256, 0, stream>>>(
        q_bf16, Waw_p, baw, nullptr, aw_log, NQ, AW_COLS, EMBED);

    // ---- 4) softmax + bilinear MSDA sampling + cam-masked aggregation ----
    sample_kernel<<<NQ, 256, 0, stream>>>(vproj, off_log, aw_log, refpts, bev_mask, slots_b16);

    // ---- 5) out = slots @ Wout + bout + query (residual) ----
    wmma_gemm_kernel<<<dim3(EMBED / 256, NQ / 32), 256, 0, stream>>>(
        slots_b16, Wout_p, bout, query, out, NQ, EMBED, EMBED);
}
